// Attention_17763984736748
// MI455X (gfx1250) — matrix operand order repaired and checked by reading, not compiled, not run on hardware
//
#include <hip/hip_runtime.h>

typedef __attribute__((ext_vector_type(16))) _Float16 v16h;
typedef __attribute__((ext_vector_type(8)))  _Float16 v8h;
typedef __attribute__((ext_vector_type(8)))  float    v8f;
typedef unsigned int u32x4 __attribute__((ext_vector_type(4)));
typedef int          i32x4 __attribute__((ext_vector_type(4)));
typedef int          i32x8 __attribute__((ext_vector_type(8)));

#if defined(__has_builtin)
#if __has_builtin(__builtin_amdgcn_tensor_load_to_lds) && \
    __has_builtin(__builtin_amdgcn_s_wait_tensorcnt)
#define ATHENA_HAS_TDM 1
#endif
#endif
#ifndef ATHENA_HAS_TDM
#define ATHENA_HAS_TDM 0
#endif

namespace {
constexpr int   kHeads  = 8;
constexpr int   kDimH   = 64;
constexpr int   kSeq    = 197;
constexpr int   kBatch  = 64;
constexpr int   kDModel = 384;
constexpr int   kInner  = 512;          // heads * dim_head
constexpr int   kNQkv   = 1536;         // 3 * inner
constexpr int   kMRows  = kBatch * kSeq;   // 12608 == 788*16 exactly
constexpr int   kMTiles = kMRows / 16;     // 788 (even -> 394 double-M blocks)
constexpr float kNegMask = -987654321.0f;

// Attention LDS layout (fits CDNA5's 320KB/WGP; one workgroup per WGP)
constexpr int    kSStride = 224;  // floats per S row (K padded: 7 * 32)
constexpr int    kPStride = 448;  // halfs  per P row (same bytes, in-place overlay)
constexpr int    kSRows   = 208;  // 197 padded to 13*16
constexpr size_t kSBytes  = (size_t)kSRows * kSStride * 4;  // 186368
constexpr size_t kQBytes  = (size_t)kSeq * kDimH * 2;       // valid q payload 25216
constexpr size_t kQRegion = (size_t)kSRows * kDimH * 2;     // 26624 (padded rows)
constexpr size_t kVRegion = (size_t)kDimH * kSStride * 2;   // 28672
constexpr size_t kRSBytes = (size_t)kSRows * 4;             // 832
constexpr size_t kSmemBytes =
    kSBytes + kQRegion + kQRegion + kVRegion + kRSBytes;    // 269120 < 320K
}

// A fragment (16x32 f16, row-major source): lane holds row (l&15);
// halfs [cb..cb+7] -> v0..3, [cb+16..cb+23] -> v4..7, cb = k0 + 8*(l>>4)
__device__ __forceinline__ v16h frag_a(const _Float16* p) {
  v8h lo = *(const v8h*)p;
  v8h hi = *(const v8h*)(p + 16);
  return __builtin_shufflevector(lo, hi, 0,1,2,3,4,5,6,7,8,9,10,11,12,13,14,15);
}
// B fragment (32x16 f16, column-major source): lane holds column (l&15);
// halfs [cb..cb+7] -> v0..3, [cb+16..cb+23] -> v4..7, cb = k0 + 8*(l>>4)
__device__ __forceinline__ v16h frag_b(const _Float16* p) {
  v8h lo = *(const v8h*)p;
  v8h hi = *(const v8h*)(p + 16);
  return __builtin_shufflevector(lo, hi, 0,1,2,3,4,5,6,7,8,9,10,11,12,13,14,15);
}

__device__ __forceinline__ v8f wmma_f16(v16h a, v16h b, v8f c) {
  return __builtin_amdgcn_wmma_f32_16x16x32_f16(false, a, false, b, (short)0, c,
                                                false, false);
}

#if ATHENA_HAS_TDM
// 1-D contiguous global->LDS DMA via the Tensor Data Mover.
// D# per ISA cdna5 ch.8: group0 = {count=1, lds_addr, global_addr, type=2},
// group1 = {data_size=8B, tensor_dim0=tile_dim0=n_qwords, dims1.. = 1}.
__device__ __forceinline__ void tdm_load_1d(const void* gsrc,
                                            unsigned lds_off_bytes,
                                            unsigned n_qwords) {
  unsigned long long ga = (unsigned long long)gsrc;
  u32x4 g0;
  g0.x = 1u;                                   // count=1, user descriptor
  g0.y = lds_off_bytes;                        // lds_addr (bytes)
  g0.z = (unsigned)(ga & 0xFFFFFFFFu);         // global_addr[31:0]
  g0.w = (unsigned)((ga >> 32) & 0x01FFFFFFu) | (2u << 30);  // [56:32] | type=2
  i32x8 g1;
  g1[0] = (3 << 16);                               // data_size = 8 bytes
  g1[1] = (int)((n_qwords & 0xFFFFu) << 16);       // tensor_dim0[15:0]
  g1[2] = (int)(((n_qwords >> 16) & 0xFFFFu) | (1u << 16)); // dim0 hi | dim1=1
  g1[3] = (int)((n_qwords & 0xFFFFu) << 16);       // tile_dim0
  g1[4] = 1;                                       // tile_dim1 = 1
  g1[5] = (int)n_qwords;                           // tensor_dim0_stride lo
  g1[6] = 0;
  g1[7] = 0;
  i32x4 z4 = {0, 0, 0, 0};
#if __clang_major__ >= 23
  i32x8 z8 = {0, 0, 0, 0, 0, 0, 0, 0};
  __builtin_amdgcn_tensor_load_to_lds(g0, g1, z4, z4, z8, 0);
#else
  __builtin_amdgcn_tensor_load_to_lds(g0, g1, z4, z4, 0);
#endif
}
#endif

// ---------------------------------------------------------------------------
// Kernel 0: convert x to f16; transpose+convert weights to "column-major B"
// ---------------------------------------------------------------------------
__global__ void pack_inputs(const float* __restrict__ x,
                            const float* __restrict__ wqkv,
                            const float* __restrict__ wout,
                            _Float16* __restrict__ xh,
                            _Float16* __restrict__ wqkvT,
                            _Float16* __restrict__ woutT) {
  const int tid = blockIdx.x * blockDim.x + threadIdx.x;
  const int stride = gridDim.x * blockDim.x;
  for (int i = tid; i < kMRows * kDModel; i += stride)
    xh[i] = (_Float16)x[i];
  for (int i = tid; i < kNQkv * kDModel; i += stride) {
    int c = i / kDModel, k = i - c * kDModel;
    wqkvT[i] = (_Float16)wqkv[k * kNQkv + c];
  }
  for (int i = tid; i < kDModel * kInner; i += stride) {
    int n = i / kInner, k = i - n * kInner;
    woutT[i] = (_Float16)wout[k * kDModel + n];
  }
}

// ---------------------------------------------------------------------------
// Kernel 1: qkv = x @ w_qkv -> q[b,h,n,d], k[b,h,n,d], vT[b,h,d,n(pad 224)]
// One wave per 32x128 output slab (2 M-tiles share B fragments -> halves
// L2 B traffic, 16 WMMAs per K-step).
// ---------------------------------------------------------------------------
__global__ __launch_bounds__(32) void qkv_gemm(const _Float16* __restrict__ xh,
                                               const _Float16* __restrict__ wqkvT,
                                               _Float16* __restrict__ qh,
                                               _Float16* __restrict__ kh,
                                               _Float16* __restrict__ vT) {
  const int lane  = threadIdx.x & 31;
  const int half  = lane >> 4;
  const int l15   = lane & 15;
  const int mbase = blockIdx.x * 32;
  const int nslab = blockIdx.y * 128;

  v8f acc0[8] = {}, acc1[8] = {};
  const _Float16* arow0 = xh + (size_t)(mbase + l15) * kDModel;
  const _Float16* arow1 = xh + (size_t)(mbase + 16 + l15) * kDModel;

  for (int k0 = 0; k0 < kDModel; k0 += 32) {
    __builtin_prefetch(arow0 + k0 + 64, 0, 1);   // global_prefetch_b8
    v16h a0 = frag_a(arow0 + k0 + half * 8);
    v16h a1 = frag_a(arow1 + k0 + half * 8);
#pragma unroll
    for (int t = 0; t < 8; ++t) {
      const _Float16* bp =
          wqkvT + (size_t)(nslab + t * 16 + l15) * kDModel + k0 + half * 8;
      v16h b = frag_b(bp);
      acc0[t] = wmma_f16(a0, b, acc0[t]);
      acc1[t] = wmma_f16(a1, b, acc1[t]);
    }
  }

#pragma unroll
  for (int t = 0; t < 8; ++t) {
    const int n = nslab + t * 16 + l15;
    const int which = n >> 9;        // 0:q 1:k 2:v
    const int h = (n >> 6) & 7;
    const int d = n & 63;
#pragma unroll
    for (int r = 0; r < 16; ++r) {
      const int row = mbase + (r & 7) + 8 * half + (r >= 8 ? 16 : 0);
      const int bb  = row / kSeq;
      const int tok = row - bb * kSeq;
      const float accv = (r < 8) ? acc0[t][r] : acc1[t][r - 8];
      const _Float16 val = (_Float16)accv;
      const size_t bh = (size_t)bb * kHeads + h;
      if (which == 0)       qh[(bh * kSeq + tok) * kDimH + d] = val;
      else if (which == 1)  kh[(bh * kSeq + tok) * kDimH + d] = val;
      else                  vT[(bh * kDimH + d) * kSStride + tok] = val;
    }
  }
}

// ---------------------------------------------------------------------------
// Kernel 2: whole-head attention in LDS. grid = B*H, block = 256 (8 waves).
// q/k/vT staged via TDM (tensor_load_to_lds + s_wait_tensorcnt) when the
// toolchain exposes it, else vectorized copies. S = q k^T * scale, diag mask,
// softmax (unnormalized exp stored in-place as f16 over the f32 S rows),
// O = P v / rowsum.
// ---------------------------------------------------------------------------
__global__ __launch_bounds__(256) void attention_kernel(
    const _Float16* __restrict__ qh, const _Float16* __restrict__ kh,
    const _Float16* __restrict__ vT, const float* __restrict__ scale,
    _Float16* __restrict__ attnO) {
  extern __shared__ char smem[];
  float*    Sbuf   = (float*)smem;                                     // 208x224
  _Float16* qs     = (_Float16*)(smem + kSBytes);                      // 208x64
  _Float16* ks     = (_Float16*)(smem + kSBytes + kQRegion);           // 208x64
  _Float16* vs     = (_Float16*)(smem + kSBytes + 2 * kQRegion);       // 64x224
  float*    rowsum = (float*)(smem + kSBytes + 2 * kQRegion + kVRegion);

  const int tid  = threadIdx.x;
  const int bhid = blockIdx.x;
  const int bb   = bhid >> 3;
  const int h    = bhid & 7;
  const float sc = scale[h];

  const _Float16* qg = qh + (size_t)bhid * kSeq * kDimH;
  const _Float16* kg = kh + (size_t)bhid * kSeq * kDimH;
  const _Float16* vg = vT + (size_t)bhid * kDimH * kSStride;  // padded stride

#if ATHENA_HAS_TDM
  // One TDM descriptor per staging tile, issued by waves 0..2.
  if (tid < 32)        tdm_load_1d(qg, (unsigned)kSBytes, kQBytes / 8);
  else if (tid < 64)   tdm_load_1d(kg, (unsigned)(kSBytes + kQRegion),
                                   kQBytes / 8);
  else if (tid < 96)   tdm_load_1d(vg, (unsigned)(kSBytes + 2 * kQRegion),
                                   kVRegion / 8);
  __builtin_amdgcn_s_wait_tensorcnt(0);
  __syncthreads();            // DMA complete & visible before pad-zeroing
#else
  for (int c = tid; c < kSeq * kDimH / 8; c += 256) {   // rows < 197 exactly
    ((v8h*)qs)[c] = ((const v8h*)qg)[c];
    ((v8h*)ks)[c] = ((const v8h*)kg)[c];
  }
  for (int c = tid; c < kDimH * kSStride / 8; c += 256)
    ((v8h*)vs)[c] = ((const v8h*)vg)[c];
  __syncthreads();
#endif
  // Zero padding: q/k rows 197..207, vT columns 197..223.
  for (int i = kSeq * kDimH + tid; i < kSRows * kDimH; i += 256) {
    qs[i] = (_Float16)0.0f;
    ks[i] = (_Float16)0.0f;
  }
  constexpr int kPadW = kSStride - kSeq;  // 27
  for (int i = tid; i < kDimH * kPadW; i += 256) {
    const int d = i / kPadW, j = kSeq + (i - d * kPadW);
    vs[d * kSStride + j] = (_Float16)0.0f;
  }
  __syncthreads();

  const int wave = tid >> 5, lane = tid & 31;
  const int half = lane >> 4, l15 = lane & 15;

  // ---- S = q k^T (13x13 tiles of 16x16, K = 64) ----
  for (int idx = wave; idx < 169; idx += 8) {
    const int tm = idx / 13, tn = idx - tm * 13;
    v8f acc = {};
#pragma unroll
    for (int k0 = 0; k0 < kDimH; k0 += 32) {
      v16h a = frag_a(qs + (tm * 16 + l15) * kDimH + k0 + half * 8);
      v16h b = frag_b(ks + (tn * 16 + l15) * kDimH + k0 + half * 8);
      acc = wmma_f16(a, b, acc);
    }
#pragma unroll
    for (int r = 0; r < 8; ++r) {
      const int i = tm * 16 + r + 8 * half;
      const int j = tn * 16 + l15;
      float v = acc[r] * sc;
      if (i == j) v = kNegMask;           // LSA diagonal mask (after scale)
      Sbuf[i * kSStride + j] = v;
    }
  }
  __syncthreads();

  // ---- softmax row-wise; store unnormalized exp as f16 in-place ----
  if (tid < kSeq) {
    const int i = tid;
    const float* srow = Sbuf + i * kSStride;
    float m = srow[0];
    for (int j = 1; j < kSeq; ++j) m = fmaxf(m, srow[j]);
    volatile float*    vrow = Sbuf + i * kSStride;
    volatile _Float16* prow = ((volatile _Float16*)Sbuf) + (size_t)i * kPStride;
    float s = 0.0f;
    for (int j = 0; j < kSeq; ++j) {      // f16 write at byte 2j never clobbers
      const float e = __expf(vrow[j] - m);  // an un-read f32 at byte 4j
      s += e;
      prow[j] = (_Float16)e;
    }
    for (int j = kSeq; j < kSStride; ++j) prow[j] = (_Float16)0.0f;  // K pad
    rowsum[i] = s;
  }
  __syncthreads();

  // ---- O = P @ v (13x4 tiles, K padded to 224), fold in 1/rowsum ----
  const _Float16* P16 = (const _Float16*)Sbuf;
  for (int idx = wave; idx < 52; idx += 8) {
    const int tm = idx >> 2, tn = idx & 3;
    v8f acc = {};
#pragma unroll
    for (int k0 = 0; k0 < kSStride; k0 += 32) {
      v16h a = frag_a(P16 + (tm * 16 + l15) * kPStride + k0 + half * 8);
      v16h b = frag_b(vs + (tn * 16 + l15) * kSStride + k0 + half * 8);
      acc = wmma_f16(a, b, acc);
    }
#pragma unroll
    for (int r = 0; r < 8; ++r) {
      const int i = tm * 16 + r + 8 * half;
      const int d = tn * 16 + l15;
      if (i < kSeq) {
        const float o = acc[r] / rowsum[i];
        attnO[((size_t)bb * kSeq + i) * kInner + h * kDimH + d] = (_Float16)o;
      }
    }
  }
}

// ---------------------------------------------------------------------------
// Kernel 3: out = attnO @ w_out + b_out (fp32 output), 2 M-tiles per wave
// ---------------------------------------------------------------------------
__global__ __launch_bounds__(32) void out_gemm(const _Float16* __restrict__ attnO,
                                               const _Float16* __restrict__ woutT,
                                               const float* __restrict__ bout,
                                               float* __restrict__ out) {
  const int lane  = threadIdx.x & 31;
  const int half  = lane >> 4;
  const int l15   = lane & 15;
  const int mbase = blockIdx.x * 32;
  const int nslab = blockIdx.y * 128;

  v8f acc0[8] = {}, acc1[8] = {};
  const _Float16* arow0 = attnO + (size_t)(mbase + l15) * kInner;
  const _Float16* arow1 = attnO + (size_t)(mbase + 16 + l15) * kInner;

  for (int k0 = 0; k0 < kInner; k0 += 32) {
    __builtin_prefetch(arow0 + k0 + 64, 0, 1);
    v16h a0 = frag_a(arow0 + k0 + half * 8);
    v16h a1 = frag_a(arow1 + k0 + half * 8);
#pragma unroll
    for (int t = 0; t < 8; ++t) {
      const _Float16* bp =
          woutT + (size_t)(nslab + t * 16 + l15) * kInner + k0 + half * 8;
      v16h b = frag_b(bp);
      acc0[t] = wmma_f16(a0, b, acc0[t]);
      acc1[t] = wmma_f16(a1, b, acc1[t]);
    }
  }

#pragma unroll
  for (int t = 0; t < 8; ++t) {
    const int n = nslab + t * 16 + l15;
    const float bias = bout[n];
#pragma unroll
    for (int r = 0; r < 8; ++r) {
      const int row0 = mbase + r + 8 * half;
      out[(size_t)row0 * kDModel + n] = acc0[t][r] + bias;
      out[(size_t)(row0 + 16) * kDModel + n] = acc1[t][r] + bias;
    }
  }
}

// ---------------------------------------------------------------------------
extern "C" void kernel_launch(void* const* d_in, const int* in_sizes, int n_in,
                              void* d_out, int out_size, void* d_ws,
                              size_t ws_size, hipStream_t stream) {
  (void)in_sizes; (void)n_in; (void)out_size; (void)ws_size;
  const float* x     = (const float*)d_in[0];
  const float* wqkv  = (const float*)d_in[1];
  const float* scale = (const float*)d_in[2];
  const float* wout  = (const float*)d_in[3];
  const float* bout  = (const float*)d_in[4];
  float* out = (float*)d_out;

  char* ws = (char*)d_ws;
  size_t off = 0;
  auto alloc = [&](size_t bytes) -> char* {
    char* p = ws + off;
    off = (off + bytes + 255) & ~(size_t)255;
    return p;
  };
  _Float16* xh    = (_Float16*)alloc((size_t)kMRows * kDModel * 2);
  _Float16* wqkvT = (_Float16*)alloc((size_t)kNQkv * kDModel * 2);
  _Float16* woutT = (_Float16*)alloc((size_t)kDModel * kInner * 2);
  _Float16* qh    = (_Float16*)alloc((size_t)kBatch * kHeads * kSeq * kDimH * 2);
  _Float16* kh    = (_Float16*)alloc((size_t)kBatch * kHeads * kSeq * kDimH * 2);
  _Float16* vT    = (_Float16*)alloc((size_t)kBatch * kHeads * kDimH * kSStride * 2);
  _Float16* attnO = (_Float16*)alloc((size_t)kMRows * kInner * 2);

  pack_inputs<<<512, 256, 0, stream>>>(x, wqkv, wout, xh, wqkvT, woutT);
  qkv_gemm<<<dim3(kMTiles / 2, kNQkv / 128), 32, 0, stream>>>(xh, wqkvT, qh, kh,
                                                              vT);
  attention_kernel<<<kBatch * kHeads, 256, kSmemBytes, stream>>>(qh, kh, vT,
                                                                 scale, attnO);
  out_gemm<<<dim3(kMTiles / 2, kDModel / 128), 32, 0, stream>>>(attnO, woutT,
                                                                bout, out);
}
